// MultiTimeframeEncoder_11355893531351
// MI455X (gfx1250) — compile-verified, matrix operand order checked by reading, not run on hardware
//
#include <hip/hip_runtime.h>

// ---------------------------------------------------------------------------
// Types for CDNA5 WMMA (wave32): v_wmma_f32_16x16x32_bf16
// ---------------------------------------------------------------------------
typedef __attribute__((ext_vector_type(16))) __bf16 v16bf;
typedef __attribute__((ext_vector_type(8)))  float  v8f;
typedef __bf16 bf16_t;

#define HID   256
#define BTOT  2048
#define ROWS  32          // batch rows per workgroup (2 row-tiles of 16)
#define NWAVE 8           // 256 threads / wave32

// ---------------------------------------------------------------------------
// helpers
// ---------------------------------------------------------------------------
__device__ __forceinline__ bf16_t f2bf(float f) { return (bf16_t)f; }  // hw cvt

__device__ __forceinline__ v16bf frag_from_bf(const bf16_t* p) {
  return *(const v16bf*)p;   // 32B aligned by construction
}
__device__ __forceinline__ float sigmoid_f(float x) { return 1.0f / (1.0f + __expf(-x)); }

__device__ __forceinline__ v8f wmma_bf16(v16bf a, v16bf b, v8f c) {
  return __builtin_amdgcn_wmma_f32_16x16x32_bf16(false, a, false, b, (short)0, c, false, false);
}

// ---------------------------------------------------------------------------
// sanitize x (NaN->0, Inf->sign*10) and convert to bf16
// ---------------------------------------------------------------------------
__global__ void k_sanitize_bf16(const float* __restrict__ x, bf16_t* __restrict__ y, long n) {
  long i = (long)blockIdx.x * blockDim.x + threadIdx.x;
  if (i >= n) return;
  float v = x[i];
  if (isnan(v)) v = 0.0f;
  else if (isinf(v)) v = (v > 0.0f) ? 10.0f : -10.0f;
  y[i] = f2bf(v);
}

// ---------------------------------------------------------------------------
// dst (ncols x K, row-major bf16) = transpose of first `ncols` columns of
// src (K x N, row-major f32).  One-time weight prep; coalescing irrelevant.
// ---------------------------------------------------------------------------
__global__ void k_transpose_bf16(const float* __restrict__ src, bf16_t* __restrict__ dst,
                                 int K, int N, int ncols) {
  long i = (long)blockIdx.x * blockDim.x + threadIdx.x;
  long tot = (long)K * ncols;
  if (i >= tot) return;
  int n = (int)(i / K);
  int k = (int)(i % K);
  dst[i] = f2bf(src[(size_t)k * N + n]);
}

// ---------------------------------------------------------------------------
// Fused 2-layer GRU, persistent over the time loop.
//   gates = [h | x] @ Wzr + bzr     (512 cols only; the reference's
//                                    jnp.split discards the third chunk)
//   cand  = [r*h | x] @ Wc + bc
//   h     = (1-z)*h + z*tanh(cand)
// All WMMA A/B operands are bf16 in LDS / L2: h keeps an f32 master copy for
// the state update plus a bf16 mirror refreshed once per step, so the K-loops
// contain no float->bf16 conversion VALU at all.
// ---------------------------------------------------------------------------
struct GruArgs {
  const bf16_t* xbf[4];
  const bf16_t* wzr0[4]; const bf16_t* wc0[4];
  const bf16_t* wzr1[4]; const bf16_t* wc1[4];
  const float*  bzr0[4]; const float*  bc0[4];
  const float*  bzr1[4]; const float*  bc1[4];
  float*        hlast[4];
  int T[4];
};

template<int KTOT>   // 288 (layer0: x width 32) or 512 (layer1: x = h0, width 256)
__device__ void gru_sublayer(float*  __restrict__ h,    // LDS [ROWS][HID] f32 master
                             bf16_t* __restrict__ hbf,  // LDS [ROWS][HID] bf16 mirror
                             const bf16_t* __restrict__ xbf, // LDS [ROWS][KTOT-256]
                             bf16_t* __restrict__ rh,   // LDS [ROWS][HID]
                             float*  __restrict__ zz,   // LDS [ROWS][HID]
                             const bf16_t* __restrict__ WzrT,  // [512][KTOT]
                             const float*  __restrict__ bzr,
                             const bf16_t* __restrict__ WcT,   // [256][KTOT]
                             const float*  __restrict__ bc,
                             int lane, int wv) {
  const int XW = KTOT - 256;
  const int m  = lane & 15;
  const int kh = (lane >> 4) << 4;     // 0 / 16 : K half
  const int hi = (lane >> 4) << 3;     // 0 / 8  : row half for C/D tiles

  // ---------------- z/r gates: (ROWS x KTOT) @ (KTOT x 512) ----------------
  v8f acc[2][4];
#pragma unroll
  for (int rt = 0; rt < 2; ++rt)
#pragma unroll
    for (int c = 0; c < 4; ++c) acc[rt][c] = (v8f){0,0,0,0,0,0,0,0};

  for (int kb = 0; kb < KTOT; kb += 32) {
    int kk = kb + kh;
    v16bf a[2];
#pragma unroll
    for (int rt = 0; rt < 2; ++rt) {
      int row = rt * 16 + m;
      a[rt] = (kk < 256) ? frag_from_bf(&hbf[row * HID + kk])
                         : frag_from_bf(&xbf[row * XW + (kk - 256)]);
    }
    v16bf b[4];
#pragma unroll
    for (int c = 0; c < 4; ++c) {
      int n = (wv * 4 + c) * 16 + m;
      b[c] = frag_from_bf(&WzrT[(size_t)n * KTOT + kk]);
    }
#pragma unroll
    for (int rt = 0; rt < 2; ++rt)
#pragma unroll
      for (int c = 0; c < 4; ++c) acc[rt][c] = wmma_bf16(a[rt], b[c], acc[rt][c]);
  }
  // epilogue: z -> zz (f32 LDS), r -> r*h (bf16 LDS)
#pragma unroll
  for (int rt = 0; rt < 2; ++rt)
#pragma unroll
    for (int c = 0; c < 4; ++c) {
      int ct  = wv * 4 + c;
      int col = ct * 16 + m;
      float bias = bzr[col];
#pragma unroll
      for (int j = 0; j < 8; ++j) {
        int row = rt * 16 + hi + j;
        float s = sigmoid_f(acc[rt][c][j] + bias);
        if (ct < 16) zz[row * HID + col] = s;
        else {
          int hc = col - 256;
          rh[row * HID + hc] = f2bf(s * h[row * HID + hc]);
        }
      }
    }
  __syncthreads();

  // ---------------- candidate: (ROWS x KTOT) @ (KTOT x 256) ----------------
  v8f cacc[2][2];
#pragma unroll
  for (int rt = 0; rt < 2; ++rt)
#pragma unroll
    for (int c = 0; c < 2; ++c) cacc[rt][c] = (v8f){0,0,0,0,0,0,0,0};

  for (int kb = 0; kb < KTOT; kb += 32) {
    int kk = kb + kh;
    v16bf a[2];
#pragma unroll
    for (int rt = 0; rt < 2; ++rt) {
      int row = rt * 16 + m;
      a[rt] = (kk < 256) ? frag_from_bf(&rh[row * HID + kk])
                         : frag_from_bf(&xbf[row * XW + (kk - 256)]);
    }
    v16bf b[2];
#pragma unroll
    for (int c = 0; c < 2; ++c) {
      int n = (wv * 2 + c) * 16 + m;
      b[c] = frag_from_bf(&WcT[(size_t)n * KTOT + kk]);
    }
#pragma unroll
    for (int rt = 0; rt < 2; ++rt)
#pragma unroll
      for (int c = 0; c < 2; ++c) cacc[rt][c] = wmma_bf16(a[rt], b[c], cacc[rt][c]);
  }
  // epilogue: h = (1-z)*h + z*tanh(cand); refresh f32 master + bf16 mirror
#pragma unroll
  for (int rt = 0; rt < 2; ++rt)
#pragma unroll
    for (int c = 0; c < 2; ++c) {
      int col = (wv * 2 + c) * 16 + m;
      float bias = bc[col];
#pragma unroll
      for (int j = 0; j < 8; ++j) {
        int row = rt * 16 + hi + j;
        float cand = tanhf(cacc[rt][c][j] + bias);
        float zv = zz[row * HID + col];
        float ho = h[row * HID + col];
        float hn = (1.0f - zv) * ho + zv * cand;
        h[row * HID + col]   = hn;
        hbf[row * HID + col] = f2bf(hn);
      }
    }
  __syncthreads();
}

extern "C" __global__ void __launch_bounds__(256)
k_gru_stack(GruArgs args) {
  extern __shared__ char smem[];
  float*  h0   = (float*)smem;                    // ROWS*HID f32
  float*  h1   = h0 + ROWS * HID;                 // ROWS*HID f32
  float*  zz   = h1 + ROWS * HID;                 // ROWS*HID f32
  bf16_t* h0bf = (bf16_t*)(zz + ROWS * HID);      // ROWS*HID bf16
  bf16_t* h1bf = h0bf + ROWS * HID;               // ROWS*HID bf16
  bf16_t* rh   = h1bf + ROWS * HID;               // ROWS*HID bf16
  bf16_t* xt   = rh + ROWS * HID;                 // ROWS*32  bf16

  const int tf      = blockIdx.y;
  const int T       = args.T[tf];
  const int rowbase = blockIdx.x * ROWS;
  const int tid     = threadIdx.x;
  const int lane    = tid & 31;
  const int wv      = tid >> 5;

  for (int i = tid; i < ROWS * HID; i += 256) {
    h0[i] = 0.0f; h1[i] = 0.0f;
    h0bf[i] = f2bf(0.0f); h1bf[i] = f2bf(0.0f);
  }
  __syncthreads();

  const bf16_t* x = args.xbf[tf];
  for (int t = 0; t < T; ++t) {
    // stage x_t tile (ROWS x 32) into LDS
    for (int i = tid; i < ROWS * 32; i += 256) {
      int r = i >> 5, c = i & 31;
      xt[i] = x[(size_t)(rowbase + r) * ((size_t)T * 32) + (size_t)t * 32 + c];
    }
    __syncthreads();
    gru_sublayer<288>(h0, h0bf, xt, rh, zz,
                      args.wzr0[tf], args.bzr0[tf], args.wc0[tf], args.bc0[tf], lane, wv);
    gru_sublayer<512>(h1, h1bf, h0bf, rh, zz,
                      args.wzr1[tf], args.bzr1[tf], args.wc1[tf], args.bc1[tf], lane, wv);
  }
  float* out = args.hlast[tf] + (size_t)rowbase * HID;
  for (int i = tid; i < ROWS * HID; i += 256) out[i] = h1[i];
}

// ---------------------------------------------------------------------------
// LayerNorm(h_last) -> bf16 into concatenated all_enc at sorted-TF col offset
// ---------------------------------------------------------------------------
struct LnArgs { const float* hlast[4]; const float* g[4]; const float* b[4]; int colofs[4]; };

extern "C" __global__ void __launch_bounds__(256)
k_layernorm(LnArgs a, bf16_t* __restrict__ allenc) {
  __shared__ float red[256];
  const int tf = blockIdx.y, row = blockIdx.x, tid = threadIdx.x;
  float v = a.hlast[tf][(size_t)row * HID + tid];
  red[tid] = v; __syncthreads();
  for (int s = 128; s > 0; s >>= 1) { if (tid < s) red[tid] += red[tid + s]; __syncthreads(); }
  float mu = red[0] / 256.0f; __syncthreads();
  float d = v - mu;
  red[tid] = d * d; __syncthreads();
  for (int s = 128; s > 0; s >>= 1) { if (tid < s) red[tid] += red[tid + s]; __syncthreads(); }
  float var = red[0] / 256.0f;
  float y = d * rsqrtf(var + 1e-5f) * a.g[tf][tid] + a.b[tf][tid];
  allenc[(size_t)row * 1024 + a.colofs[tf] + tid] = f2bf(y);
}

// ---------------------------------------------------------------------------
// Generic WMMA GEMM: C(MxN) = A(MxK,bf16) @ BT(NxK,bf16)^T + bias
// one 16x16 tile per wave; A/B fragments straight from L2-resident operands.
// ---------------------------------------------------------------------------
template<bool OUTF32>
__global__ void __launch_bounds__(256)
k_gemm_bf16(const bf16_t* __restrict__ A, const bf16_t* __restrict__ BT,
            const float* __restrict__ bias, void* __restrict__ C,
            int M, int N, int K) {
  const int lane = threadIdx.x & 31, wv = threadIdx.x >> 5;
  const int tile = blockIdx.x * NWAVE + wv;
  const int ntiles = (M / 16) * (N / 16);
  if (tile >= ntiles) return;
  const int mt = tile / (N / 16), nt = tile % (N / 16);
  const int m = lane & 15, kh = (lane >> 4) << 4, hi = (lane >> 4) << 3;

  v8f acc = (v8f){0,0,0,0,0,0,0,0};
  const bf16_t* arow = A  + (size_t)(mt * 16 + m) * K;
  const bf16_t* brow = BT + (size_t)(nt * 16 + m) * K;
  for (int kb = 0; kb < K; kb += 32)
    acc = wmma_bf16(frag_from_bf(arow + kb + kh), frag_from_bf(brow + kb + kh), acc);

  const int col = nt * 16 + m;
  const float bs = bias ? bias[col] : 0.0f;
#pragma unroll
  for (int j = 0; j < 8; ++j) {
    int row = mt * 16 + hi + j;
    float v = acc[j] + bs;
    if (OUTF32) ((float*)C)[(size_t)row * N + col] = v;
    else        ((bf16_t*)C)[(size_t)row * N + col] = f2bf(v);
  }
}

// ---------------------------------------------------------------------------
// host launcher
// ---------------------------------------------------------------------------
extern "C" void kernel_launch(void* const* d_in, const int* in_sizes, int n_in,
                              void* d_out, int out_size, void* d_ws, size_t ws_size,
                              hipStream_t stream) {
  (void)in_sizes; (void)n_in; (void)out_size; (void)ws_size;
  // TF order here: 0=M1,1=M5,2=M15,3=H1 (matches x input order)
  const int Ts[4] = {240, 96, 64, 48};
  // params flattened after the 4 x inputs in jax pytree (sorted-key) order:
  // tf blocks sorted: H1@4, M1@14, M15@24, M5@34; attn@44.
  // per tf block: +0 L0.cand.b +1 L0.cand.w +2 L0.zrh.b +3 L0.zrh.w
  //               +4 L1.cand.b +5 L1.cand.w +6 L1.zrh.b +7 L1.zrh.w +8 ln_b +9 ln_g
  const int pb[4] = {14, 34, 24, 4};      // M1, M5, M15, H1
  const int ab = 44;                      // attn: +0 k.b +1 k.w +2 o.b +3 o.w +4 q.b +5 q.w +6 v.b +7 v.w
  // sorted(TFS) concat order = [H1, M1, M15, M5] -> column offsets per our tf idx
  const int colofs[4] = {256, 768, 512, 0};

  size_t off = 0;
  auto carve = [&](size_t bytes) -> char* {
    char* p = (char*)d_ws + off;
    off += (bytes + 255) & ~(size_t)255;
    return p;
  };

  bf16_t* xbf[4]; bf16_t* wzr0[4]; bf16_t* wc0[4]; bf16_t* wzr1[4]; bf16_t* wc1[4];
  float* hlast[4];
  for (int tf = 0; tf < 4; ++tf) {
    xbf[tf]  = (bf16_t*)carve((size_t)BTOT * Ts[tf] * 32 * sizeof(bf16_t));
    wzr0[tf] = (bf16_t*)carve((size_t)512 * 288 * sizeof(bf16_t));
    wc0[tf]  = (bf16_t*)carve((size_t)256 * 288 * sizeof(bf16_t));
    wzr1[tf] = (bf16_t*)carve((size_t)512 * 512 * sizeof(bf16_t));
    wc1[tf]  = (bf16_t*)carve((size_t)256 * 512 * sizeof(bf16_t));
    hlast[tf] = (float*)carve((size_t)BTOT * HID * sizeof(float));
  }
  bf16_t* allenc = (bf16_t*)carve((size_t)BTOT * 1024 * sizeof(bf16_t));
  bf16_t* vbf    = (bf16_t*)carve((size_t)BTOT * 256  * sizeof(bf16_t));
  bf16_t* WvT    = (bf16_t*)carve((size_t)256 * 1024 * sizeof(bf16_t));
  bf16_t* WoT    = (bf16_t*)carve((size_t)256 * 256  * sizeof(bf16_t));

  // 1) sanitize + convert x
  for (int tf = 0; tf < 4; ++tf) {
    long n = (long)BTOT * Ts[tf] * 32;
    k_sanitize_bf16<<<(int)((n + 255) / 256), 256, 0, stream>>>((const float*)d_in[tf], xbf[tf], n);
  }
  // 2) transpose+convert weights (only first 512 gate columns kept)
  for (int tf = 0; tf < 4; ++tf) {
    const int b = pb[tf];
    long n;
    n = (long)288 * 512;
    k_transpose_bf16<<<(int)((n + 255) / 256), 256, 0, stream>>>((const float*)d_in[b + 3], wzr0[tf], 288, 768, 512);
    n = (long)288 * 256;
    k_transpose_bf16<<<(int)((n + 255) / 256), 256, 0, stream>>>((const float*)d_in[b + 1], wc0[tf], 288, 256, 256);
    n = (long)512 * 512;
    k_transpose_bf16<<<(int)((n + 255) / 256), 256, 0, stream>>>((const float*)d_in[b + 7], wzr1[tf], 512, 768, 512);
    n = (long)512 * 256;
    k_transpose_bf16<<<(int)((n + 255) / 256), 256, 0, stream>>>((const float*)d_in[b + 5], wc1[tf], 512, 256, 256);
  }
  { long n = (long)1024 * 256;
    k_transpose_bf16<<<(int)((n + 255) / 256), 256, 0, stream>>>((const float*)d_in[ab + 7], WvT, 1024, 256, 256);
    n = (long)256 * 256;
    k_transpose_bf16<<<(int)((n + 255) / 256), 256, 0, stream>>>((const float*)d_in[ab + 3], WoT, 256, 256, 256);
  }
  // 3) fused 2-layer GRU over all 4 timeframes concurrently
  GruArgs ga;
  for (int tf = 0; tf < 4; ++tf) {
    const int b = pb[tf];
    ga.xbf[tf] = xbf[tf];
    ga.wzr0[tf] = wzr0[tf]; ga.wc0[tf] = wc0[tf];
    ga.wzr1[tf] = wzr1[tf]; ga.wc1[tf] = wc1[tf];
    ga.bzr0[tf] = (const float*)d_in[b + 2];
    ga.bc0[tf]  = (const float*)d_in[b + 0];
    ga.bzr1[tf] = (const float*)d_in[b + 6];
    ga.bc1[tf]  = (const float*)d_in[b + 4];
    ga.hlast[tf] = hlast[tf];
    ga.T[tf] = Ts[tf];
  }
  const size_t smem = (size_t)ROWS * HID * sizeof(float) * 3     // h0,h1,z (f32)
                    + (size_t)ROWS * HID * sizeof(bf16_t) * 3    // h0bf,h1bf,rh
                    + (size_t)ROWS * 32  * sizeof(bf16_t);       // x_t tile
  k_gru_stack<<<dim3(BTOT / ROWS, 4), dim3(256), smem, stream>>>(ga);

  // 4) LayerNorm into concatenated all_enc (bf16)
  LnArgs la;
  for (int tf = 0; tf < 4; ++tf) {
    const int b = pb[tf];
    la.hlast[tf] = hlast[tf];
    la.g[tf] = (const float*)d_in[b + 9];
    la.b[tf] = (const float*)d_in[b + 8];
    la.colofs[tf] = colofs[tf];
  }
  k_layernorm<<<dim3(BTOT, 4), dim3(256), 0, stream>>>(la, allenc);

  // 5) softmax over size-1 axis == 1  =>  out = (all_enc @ Wv + bv) @ Wo + bo
  {
    int ntiles = (BTOT / 16) * (256 / 16);
    int blocks = (ntiles + NWAVE - 1) / NWAVE;
    k_gemm_bf16<false><<<blocks, 256, 0, stream>>>(allenc, WvT, (const float*)d_in[ab + 6],
                                                   vbf, BTOT, 256, 1024);
    k_gemm_bf16<true><<<blocks, 256, 0, stream>>>(vbf, WoT, (const float*)d_in[ab + 2],
                                                  d_out, BTOT, 256, 256);
  }
}
